// MultiHeadAttention_32272384262773
// MI455X (gfx1250) — matrix operand order repaired and checked by reading, not compiled, not run on hardware
//
#include <hip/hip_runtime.h>

typedef __attribute__((ext_vector_type(16))) __bf16 v16bf;
typedef __attribute__((ext_vector_type(4)))  __bf16 v4bf;
typedef __attribute__((ext_vector_type(8)))  float  v8f;

#define EMB   768
#define HEADS 8
#define HDIM  96
#define NB    2
#define SEQ   4096
#define KB    64            // keys per staged block == softmax slab
#define WAVES 4
#define QB    (16 * WAVES)  // 64 queries per block

#define KS_STR 104          // padded row stride of K bf16 tile   [key][d]
#define VT_STR 72           // padded row stride of V^T bf16 tile [d][key]
#define P_STR  72           // padded row stride of P tile        [row][key]

// A-matrix (16x32 bf16) per-lane element -> K index (contiguous: 16*half + e)
__device__ __forceinline__ int a_kmap(int e, int half) {
    return half * 16 + e;
}

// max-reduction confined to a 16-lane half (C-fragment rows live in one half)
__device__ __forceinline__ float half_max(float x) {
    #pragma unroll
    for (int m = 1; m <= 8; m <<= 1) x = fmaxf(x, __shfl_xor(x, m, 32));
    return x;
}

// ---------------------------------------------------------------------------
// Kernel 1: flash attention, one (n, h, 64-query tile) per 4-wave block.
// Async (ASYNCcnt) double-buffered f32 staging -> bf16 working tiles.
// Softmax runs in base-2 with scale*log2(e) pre-folded into Q.
// ---------------------------------------------------------------------------
__global__ __launch_bounds__(128) void attn_flash_kernel(
    const float* __restrict__ Vg, const float* __restrict__ Kg,
    const float* __restrict__ Qg, __bf16* __restrict__ Xg) {

    __shared__ float  stK[2][KB * HDIM];       // 48 KB  f32 staging (async dst)
    __shared__ float  stV[2][KB * HDIM];       // 48 KB
    __shared__ __bf16 Ksm[KB * KS_STR];        // 13 KB  [key][d]
    __shared__ __bf16 Vtm[HDIM * VT_STR];      // 13.5KB [d][key] (transposed)
    __shared__ __bf16 Ps[WAVES][16 * P_STR];   // 9 KB   per-wave P staging

    const int tid  = threadIdx.x;
    const int wave = tid >> 5;
    const int lane = tid & 31;
    const int half = lane >> 4;
    const int l16  = lane & 15;

    const int h = blockIdx.y;
    const int n = blockIdx.z;
    // (1/sqrt(96)) * log2(e): base-2 softmax, folded into Q once
    const float qscale = 0.10206207261596575f * 1.4426950408889634f;

    const size_t nh = (size_t)n * SEQ * EMB + (size_t)h * HDIM;
    const float* Qp = Qg + nh;   // [q][d], row stride EMB
    const float* Kp = Kg + nh;
    const float* Vp = Vg + nh;

    const int q0 = blockIdx.x * QB + wave * 16;

    // ---- async stage of one 64-key block (12 x 16B chunks per thread/array)
    auto stage = [&](int kb, int buf) {
        const float* gK = Kp + (size_t)kb * KB * EMB;
        const float* gV = Vp + (size_t)kb * KB * EMB;
        #pragma unroll
        for (int i = 0; i < 12; ++i) {
            int ch  = tid + 128 * i;             // chunk id 0..1535
            int row = ch / 24, c4 = (ch % 24) * 4;
            const float* gk = gK + (size_t)row * EMB + c4;
            const float* gv = gV + (size_t)row * EMB + c4;
            unsigned lk = (unsigned)(uintptr_t)&stK[buf][ch * 4];
            unsigned lv = (unsigned)(uintptr_t)&stV[buf][ch * 4];
            asm volatile("global_load_async_to_lds_b128 %0, %1, off"
                         :: "v"(lk), "v"(gk) : "memory");
            asm volatile("global_load_async_to_lds_b128 %0, %1, off"
                         :: "v"(lv), "v"(gv) : "memory");
        }
    };
    // ---- convert exactly the chunks this thread async-loaded
    auto convert = [&](int buf) {
        #pragma unroll
        for (int i = 0; i < 12; ++i) {
            int ch  = tid + 128 * i;
            int row = ch / 24, c4 = (ch % 24) * 4;
            const float4 fk = *(const float4*)&stK[buf][ch * 4];
            const float4 fv = *(const float4*)&stV[buf][ch * 4];
            v4bf bk = {(__bf16)fk.x, (__bf16)fk.y, (__bf16)fk.z, (__bf16)fk.w};
            *(v4bf*)&Ksm[row * KS_STR + c4] = bk;           // row-major K
            Vtm[(c4 + 0) * VT_STR + row] = (__bf16)fv.x;    // transposed V
            Vtm[(c4 + 1) * VT_STR + row] = (__bf16)fv.y;
            Vtm[(c4 + 2) * VT_STR + row] = (__bf16)fv.z;
            Vtm[(c4 + 3) * VT_STR + row] = (__bf16)fv.w;
        }
    };

    // ---- Q strip: 16 rows x 96 dims -> 3 bf16 A fragments, pre-scaled
    v16bf qfrag[3];
    #pragma unroll
    for (int kc = 0; kc < 3; ++kc)
        #pragma unroll
        for (int e = 0; e < 16; ++e)
            qfrag[kc][e] = (__bf16)(qscale *
                Qp[(size_t)(q0 + l16) * EMB + kc * 32 + a_kmap(e, half)]);

    v16bf ones;
    #pragma unroll
    for (int e = 0; e < 16; ++e) ones[e] = (__bf16)1.0f;

    float mrow[8];
    v8f acc[7];                     // acc[0..5] = O (16x96), acc[6] = row sums
    #pragma unroll
    for (int r = 0; r < 8; ++r) mrow[r] = -1e30f;
    #pragma unroll
    for (int a = 0; a < 7; ++a)
        #pragma unroll
        for (int r = 0; r < 8; ++r) acc[a][r] = 0.f;

    const int NBLK = SEQ / KB;      // 64
    // pipeline prologue: block0 staged+converted, block1 in flight
    stage(0, 0);
    asm volatile("s_wait_asynccnt 0x0" ::: "memory");
    convert(0);
    stage(1, 1);
    __syncthreads();

    for (int kb = 0; kb < NBLK; ++kb) {
        // ================= compute one 64-key slab =================
        __bf16* Pw = &Ps[wave][0];

        // scores (already in log2 domain): four 16x16 f32 fragments
        v8f s[4];
        #pragma unroll
        for (int f = 0; f < 4; ++f) {
            v8f c;
            #pragma unroll
            for (int r = 0; r < 8; ++r) c[r] = 0.f;
            #pragma unroll
            for (int kc = 0; kc < 3; ++kc) {
                v16bf b;   // B[d][key]: row = kc*32 + 16*half + e, col = l16
                #pragma unroll
                for (int e = 0; e < 16; ++e)
                    b[e] = Ksm[(f * 16 + l16) * KS_STR + kc * 32 + half * 16 + e];
                c = __builtin_amdgcn_wmma_f32_16x16x32_bf16(
                        false, qfrag[kc], false, b, (short)0, c, false, false);
            }
            s[f] = c;
        }

        // online base-2 softmax over 64 keys (denominator via ones-WMMA)
        float alpha[8];
        #pragma unroll
        for (int r = 0; r < 8; ++r) {
            float rm = fmaxf(fmaxf(s[0][r], s[1][r]), fmaxf(s[2][r], s[3][r]));
            rm = half_max(rm);
            float mnew = fmaxf(mrow[r], rm);
            alpha[r] = __builtin_exp2f(mrow[r] - mnew);
            mrow[r]  = mnew;
            #pragma unroll
            for (int f = 0; f < 4; ++f)
                Pw[(half * 8 + r) * P_STR + f * 16 + l16] =
                    (__bf16)__builtin_exp2f(s[f][r] - mnew);
        }
        #pragma unroll
        for (int a = 0; a < 7; ++a)
            #pragma unroll
            for (int r = 0; r < 8; ++r) acc[a][r] *= alpha[r];

        // O += P (16x64) V (64x96); denominator += P * 1
        #pragma unroll
        for (int kt = 0; kt < 2; ++kt) {
            v16bf pfrag;
            #pragma unroll
            for (int e = 0; e < 16; ++e)
                pfrag[e] = Pw[l16 * P_STR + kt * 32 + a_kmap(e, half)];
            #pragma unroll
            for (int dt = 0; dt < 6; ++dt) {
                v16bf b;   // V^T: row d = dt*16 + l16, K = kt*32 + 16*half + e
                #pragma unroll
                for (int e = 0; e < 16; ++e)
                    b[e] = Vtm[(dt * 16 + l16) * VT_STR + kt * 32 + half * 16 + e];
                acc[dt] = __builtin_amdgcn_wmma_f32_16x16x32_bf16(
                              false, pfrag, false, b, (short)0, acc[dt], false, false);
            }
            acc[6] = __builtin_amdgcn_wmma_f32_16x16x32_bf16(
                         false, pfrag, false, ones, (short)0, acc[6], false, false);
        }
        // ================= pipeline bookkeeping =================
        __syncthreads();
        if (kb + 1 < NBLK) {
            asm volatile("s_wait_asynccnt 0x0" ::: "memory"); // block kb+1 landed
            if (kb + 2 < NBLK) stage(kb + 2, kb & 1);         // overlap next load
            convert((kb + 1) & 1);
            __syncthreads();
        }
    }

    // normalize (denominator = acc[6], replicated across the 16-lane half)
    #pragma unroll
    for (int r = 0; r < 8; ++r) {
        float inv = 1.0f / acc[6][r];
        int q = q0 + half * 8 + r;
        #pragma unroll
        for (int dt = 0; dt < 6; ++dt)
            Xg[((size_t)n * SEQ + q) * EMB + h * HDIM + dt * 16 + l16] =
                (__bf16)(acc[dt][r] * inv);
    }
}

// ---------------------------------------------------------------------------
// Kernel 2: out = x @ fc_w^T + fc_b.  M = 8192, K = 768, N = 768.
// Same async double-buffered staging; W tile transposed [col][k].
// ---------------------------------------------------------------------------
#define WT_STR 40
__global__ __launch_bounds__(128) void fc_gemm_kernel(
    const __bf16* __restrict__ Xg, const float* __restrict__ Wg,
    const float* __restrict__ Bg, float* __restrict__ Og) {

    __shared__ float  stW[2][32 * 64];   // 16 KB f32 staging (async dst)
    __shared__ __bf16 Wt[64 * WT_STR];   // [col][k], 5 KB

    const int tid  = threadIdx.x;
    const int wave = tid >> 5;
    const int lane = tid & 31;
    const int half = lane >> 4;
    const int l16  = lane & 15;

    const int row0 = blockIdx.x * 64 + wave * 16;
    const int col0 = blockIdx.y * 64;
    const int NK   = EMB / 32;           // 24 k-steps

    auto stageW = [&](int kb, int buf) {
        #pragma unroll
        for (int i = 0; i < 4; ++i) {
            int ch = tid + 128 * i;                 // chunk id 0..511
            int j = ch >> 3, i4 = (ch & 7) * 4;
            const float* g = Wg + (size_t)(col0 + j) * EMB + kb * 32 + i4;
            unsigned l = (unsigned)(uintptr_t)&stW[buf][ch * 4];
            asm volatile("global_load_async_to_lds_b128 %0, %1, off"
                         :: "v"(l), "v"(g) : "memory");
        }
    };
    auto convW = [&](int buf) {
        #pragma unroll
        for (int i = 0; i < 4; ++i) {
            int ch = tid + 128 * i;
            int j = ch >> 3, i4 = (ch & 7) * 4;
            const float4 f = *(const float4*)&stW[buf][ch * 4];
            v4bf b4 = {(__bf16)f.x, (__bf16)f.y, (__bf16)f.z, (__bf16)f.w};
            *(v4bf*)&Wt[j * WT_STR + i4] = b4;
        }
    };

    v8f acc[4];
    #pragma unroll
    for (int nt = 0; nt < 4; ++nt)
        #pragma unroll
        for (int r = 0; r < 8; ++r) acc[nt][r] = 0.f;

    stageW(0, 0);
    asm volatile("s_wait_asynccnt 0x0" ::: "memory");
    convW(0);
    stageW(1, 1);
    __syncthreads();

    for (int kb = 0; kb < NK; ++kb) {
        v16bf a;   // x strip, contiguous 8-element chunks -> global b128 loads
        #pragma unroll
        for (int e = 0; e < 16; ++e)
            a[e] = Xg[(size_t)(row0 + l16) * EMB + kb * 32 + a_kmap(e, half)];

        #pragma unroll
        for (int nt = 0; nt < 4; ++nt) {
            v16bf b;   // row K = 16*half + e, col = nt*16 + l16
            #pragma unroll
            for (int e = 0; e < 16; ++e)
                b[e] = Wt[(nt * 16 + l16) * WT_STR + half * 16 + e];
            acc[nt] = __builtin_amdgcn_wmma_f32_16x16x32_bf16(
                          false, a, false, b, (short)0, acc[nt], false, false);
        }

        __syncthreads();
        if (kb + 1 < NK) {
            asm volatile("s_wait_asynccnt 0x0" ::: "memory");
            if (kb + 2 < NK) stageW(kb + 2, kb & 1);
            convW((kb + 1) & 1);
            __syncthreads();
        }
    }

    #pragma unroll
    for (int nt = 0; nt < 4; ++nt) {
        float bias = Bg[col0 + nt * 16 + l16];
        #pragma unroll
        for (int r = 0; r < 8; ++r) {
            int row = row0 + half * 8 + r;
            Og[(size_t)row * EMB + col0 + nt * 16 + l16] = acc[nt][r] + bias;
        }
    }
}

extern "C" void kernel_launch(void* const* d_in, const int* in_sizes, int n_in,
                              void* d_out, int out_size, void* d_ws, size_t ws_size,
                              hipStream_t stream) {
    (void)in_sizes; (void)n_in; (void)out_size; (void)ws_size;
    const float* Vg = (const float*)d_in[0];
    const float* Kg = (const float*)d_in[1];
    const float* Qg = (const float*)d_in[2];
    const float* Wg = (const float*)d_in[3];
    const float* Bg = (const float*)d_in[4];

    __bf16* X = (__bf16*)d_ws;   // N*S*EMB bf16 = 12.6 MB scratch

    dim3 g1(SEQ / QB, HEADS, NB), b1(128);
    attn_flash_kernel<<<g1, b1, 0, stream>>>(Vg, Kg, Qg, X);

    dim3 g2((NB * SEQ) / 64, EMB / 64), b2(128);
    fc_gemm_kernel<<<g2, b2, 0, stream>>>(X, Wg, Bg, (float*)d_out);
}
